// SoftDTWLoss_20220706029789
// MI455X (gfx1250) — matrix operand order repaired and checked by reading, not compiled, not run on hardware
//
#include <hip/hip_runtime.h>
#include <hip/hip_bf16.h>

typedef __attribute__((ext_vector_type(16))) _Float16 v16h;
typedef __attribute__((ext_vector_type(8)))  _Float16 v8h;
typedef __attribute__((ext_vector_type(4)))  _Float16 v4h;
typedef __attribute__((ext_vector_type(8)))  float    v8f;

#define BATCH 32
#define TLEN  512
#define DFEAT 128

// ---------------------------------------------------------------------------
// Kernel 1: row-normalize x and y (f32 -> unit-norm f16), one wave32 per row.
// ---------------------------------------------------------------------------
__global__ __launch_bounds__(256) void normalize_rows(
    const float* __restrict__ x, const float* __restrict__ y,
    _Float16* __restrict__ xh, _Float16* __restrict__ yh) {
  const int gwave = (blockIdx.x * blockDim.x + threadIdx.x) >> 5;
  const int lane  = threadIdx.x & 31;
  const int rows  = BATCH * TLEN;           // 16384 rows per tensor

  const float* src;
  _Float16*    dst;
  int row;
  if (gwave < rows) { src = x; dst = xh; row = gwave; }
  else              { src = y; dst = yh; row = gwave - rows; }

  const float4 v = *(const float4*)(src + (size_t)row * DFEAT + lane * 4);
  float ss = v.x * v.x + v.y * v.y + v.z * v.z + v.w * v.w;
  #pragma unroll
  for (int off = 16; off > 0; off >>= 1) ss += __shfl_xor(ss, off, 32);

  const float inv = 1.0f / fmaxf(sqrtf(ss), 1e-12f);
  v4h o;
  o[0] = (_Float16)(v.x * inv);
  o[1] = (_Float16)(v.y * inv);
  o[2] = (_Float16)(v.z * inv);
  o[3] = (_Float16)(v.w * inv);
  *(v4h*)(dst + (size_t)row * DFEAT + lane * 4) = o;
}

// ---------------------------------------------------------------------------
// Kernel 2: D[b,t,s] = 1 - xn[b,t].yn[b,s] via v_wmma_f32_16x16x32_f16.
// One wave32 computes a 16(M) x 64(N) strip = 4 C tiles, 16 WMMAs total.
// Per K-step: A fragment + ALL four B fragments are materialized first
// (distinct registers) so the loads form one clause and the 4 WMMAs issue
// back-to-back instead of stalling on s_wait_loadcnt per tile.
// ---------------------------------------------------------------------------
__global__ __launch_bounds__(256) void cosdist_wmma(
    const _Float16* __restrict__ xh, const _Float16* __restrict__ yh,
    float* __restrict__ D) {
  const int wave = (blockIdx.x * blockDim.x + threadIdx.x) >> 5;
  const int lane = threadIdx.x & 31;
  const int l15  = lane & 15;
  const bool hi  = lane >= 16;

  int t = wave;
  const int ng = t % (TLEN / 64); t /= (TLEN / 64);   // n-strip: 4 tiles
  const int tm = t % (TLEN / 16); t /= (TLEN / 16);   // m tile
  const int b  = t;                                   // batch

  const _Float16* arow =
      xh + ((size_t)(b * TLEN + tm * 16 + l15)) * DFEAT + (hi ? 8 : 0);
  const _Float16* brow0 =
      yh + ((size_t)(b * TLEN + ng * 64 + l15)) * DFEAT + (hi ? 8 : 0);

  v8f acc[4] = {{}, {}, {}, {}};
  #pragma unroll
  for (int k0 = 0; k0 < DFEAT; k0 += 32) {
    union { v16h v; v8h h[2]; } a;
    union { v16h v; v8h h[2]; } bb[4];
    // Gather all operands for this k-step first (10 x b128 clause).
    a.h[0] = *(const v8h*)(arow + k0);
    a.h[1] = *(const v8h*)(arow + k0 + 16);
    #pragma unroll
    for (int q = 0; q < 4; ++q) {
      const _Float16* brow = brow0 + (size_t)q * 16 * DFEAT;
      bb[q].h[0] = *(const v8h*)(brow + k0);
      bb[q].h[1] = *(const v8h*)(brow + k0 + 16);
    }
    // Then issue the 4 matrix ops back-to-back.
    #pragma unroll
    for (int q = 0; q < 4; ++q) {
      acc[q] = __builtin_amdgcn_wmma_f32_16x16x32_f16(
          false, a.v, false, bb[q].v, (short)0, acc[q], false, false);
    }
  }

  // C/D layout: VGPR r -> M = r + (hi ? 8 : 0); N = l15.
  #pragma unroll
  for (int q = 0; q < 4; ++q) {
    float* out = D + (((size_t)b * TLEN + tm * 16 + (hi ? 8 : 0)) * TLEN)
                   + (ng * 64 + q * 16) + l15;
    #pragma unroll
    for (int r = 0; r < 8; ++r) out[(size_t)r * TLEN] = 1.0f - acc[q][r];
  }
}

// ---------------------------------------------------------------------------
// Kernel 3: soft-DTW wavefront. One block per batch, 512 threads (16 waves).
// R[i][j] = D[i-1][j-1] + softmin(R[i-1][j-1], R[i][j-1], R[i-1][j]).
// Three rotating LDS diagonal buffers; one barrier per anti-diagonal.
// Cost row for diagonal k+1 is double-buffered into LDS with
// GLOBAL_LOAD_ASYNC_TO_LDS_B32 (ASYNCcnt) while diagonal k is computed;
// each lane stages & consumes only its own slot -> s_wait_asynccnt suffices.
// ---------------------------------------------------------------------------
__global__ __launch_bounds__(512) void softdtw_wavefront(
    const float* __restrict__ D, float* __restrict__ out) {
  __shared__ float bufA[TLEN + 1], bufB[TLEN + 1], bufC[TLEN + 1];
  __shared__ float costbuf[2][TLEN];
  const int b   = blockIdx.x;
  const int tid = threadIdx.x;
  const int i   = tid + 1;                  // row index 1..512
  const float INF = __builtin_inff();

  float* prev2 = bufA;                      // diagonal k-2
  float* prev  = bufB;                      // diagonal k-1
  float* cur   = bufC;                      // diagonal k

  bufA[i] = INF;
  bufB[i] = INF;
  if (tid == 0) { bufA[0] = 0.0f; bufB[0] = INF; bufC[0] = INF; }

  const float* Db = D + (size_t)b * TLEN * TLEN;
  // 32-bit LDS byte offsets (flat LDS-aperture addr truncates to LDS offset)
  const unsigned cOff0 = (unsigned)(uintptr_t)&costbuf[0][tid];
  const unsigned cOff1 = (unsigned)(uintptr_t)&costbuf[1][tid];

  // Prologue: stage diagonal k=2 into costbuf[0] (clamped addr, full EXEC).
  {
    int j  = 2 - i;
    int jc = j < 1 ? 1 : (j > TLEN ? TLEN : j);
    unsigned long long ga =
        (unsigned long long)(uintptr_t)(Db + (size_t)(i - 1) * TLEN + (jc - 1));
    asm volatile("global_load_async_to_lds_b32 %0, %1, off"
                 :: "v"(cOff0), "v"(ga) : "memory");
  }
  __syncthreads();

  for (int k = 2; k <= 2 * TLEN; ++k) {
    // Issue async staging of diagonal k+1 while we consume diagonal k.
    {
      int jn = k + 1 - i;
      int jc = jn < 1 ? 1 : (jn > TLEN ? TLEN : jn);
      unsigned long long ga =
          (unsigned long long)(uintptr_t)(Db + (size_t)(i - 1) * TLEN + (jc - 1));
      unsigned dstOff = ((k + 1) & 1) ? cOff1 : cOff0;
      asm volatile("global_load_async_to_lds_b32 %0, %1, off"
                   :: "v"(dstOff), "v"(ga) : "memory");
    }
    // Wait for diagonal k's staging; leave the just-issued load in flight.
    asm volatile("s_wait_asynccnt 0x1" ::: "memory");

    const int j = k - i;
    float val = INF;
    if (j >= 1 && j <= TLEN) {
      const float cost = costbuf[k & 1][tid];
      const float rd = prev2[i - 1];        // R[i-1][j-1]
      const float rl = prev[i];             // R[i][j-1]
      const float ru = prev[i - 1];         // R[i-1][j]
      const float m  = fminf(fminf(rd, rl), ru);
      if (m < 3.0e38f) {                    // not all-inf
        const float s = __expf(m - rd) + __expf(m - rl) + __expf(m - ru);
        val = cost + m - __logf(s);
      }
    }
    cur[i] = val;
    if (tid == 0) cur[0] = INF;             // R[0][k] border
    __syncthreads();
    float* t = prev2; prev2 = prev; prev = cur; cur = t;
  }

  // Final diagonal (k = 2T) now lives in `prev`; R[T][T] at index T.
  if (i == TLEN) out[b] = prev[TLEN];
}

// ---------------------------------------------------------------------------
// Launch: ws layout = [ xh: 4MB | yh: 4MB | D: 32MB ]  (40MB total)
// ---------------------------------------------------------------------------
extern "C" void kernel_launch(void* const* d_in, const int* in_sizes, int n_in,
                              void* d_out, int out_size, void* d_ws, size_t ws_size,
                              hipStream_t stream) {
  const float* x = (const float*)d_in[0];
  const float* y = (const float*)d_in[1];
  float* out = (float*)d_out;

  char* ws = (char*)d_ws;
  _Float16* xh = (_Float16*)ws;
  _Float16* yh = (_Float16*)(ws + (size_t)BATCH * TLEN * DFEAT * sizeof(_Float16));
  float*    Dm = (float*)   (ws + (size_t)2 * BATCH * TLEN * DFEAT * sizeof(_Float16));

  // 1) normalize: 2 * 32 * 512 rows, one wave each, 8 waves/block
  {
    const int waves  = 2 * BATCH * TLEN;    // 32768
    const int blocks = waves / 8;           // 4096
    normalize_rows<<<blocks, 256, 0, stream>>>(x, y, xh, yh);
  }
  // 2) WMMA cosine-distance GEMM: 32 batches * 32 m-tiles * 8 n-strips waves
  {
    const int waves  = BATCH * (TLEN / 16) * (TLEN / 64);  // 8192
    const int blocks = waves / 8;                           // 1024
    cosdist_wmma<<<blocks, 256, 0, stream>>>(xh, yh, Dm);
  }
  // 3) soft-DTW wavefront: one block per batch
  softdtw_wavefront<<<BATCH, TLEN, 0, stream>>>(Dm, out);
}
